// HSumGraph_33251636805750
// MI455X (gfx1250) — matrix operand order repaired and checked by reading, not compiled, not run on hardware
//
#include <hip/hip_runtime.h>
#include <hip/hip_bf16.h>
#include <math.h>

// ---------------- model dimensions ----------------
#define Bc      32
#define Pc      8
#define Kc      16
#define S_DOCc  128
#define NSc     4096
#define NWc     131072
#define Lc      32
#define E_WSc   524288
#define E_WPc   131072
#define Dc      300
#define Hc      128
#define NFc     128
#define LHc     128
#define FFNc    512
#define FEc     10

typedef _Float16 v16h __attribute__((ext_vector_type(16)));
typedef float    v8f  __attribute__((ext_vector_type(8)));
typedef _Float16 h2v  __attribute__((ext_vector_type(2)));

#define DEV __device__ __forceinline__

DEV int koffA(int j, int half) { return 2*j + 16*half; }
DEV int koffB(int j, int half) { return 2*j + 16*half; }

DEV float dev_sigmoid(float x) { return 1.0f / (1.0f + __expf(-x)); }

DEV void atomicMaxF(float* addr, float v) {
  if (v >= 0.0f) atomicMax((int*)addr, __float_as_int(v));
  else           atomicMin((unsigned int*)addr, __float_as_uint(v));
}

// ---------------- generic WMMA GEMM: C = act(A@W + bias) ----------------
// A: MxK f32 row-major, W: KxNN f32 row-major. NN is compile-time so all
// B-fragment loads become immediate-offset loads from one base register.
// One wave computes a 16x64 strip: A fragment reused across 4 independent WMMAs.
// REQUIRES: M % 16 == 0 and K even (true at all call sites).
template<int NN, bool KPAD>
__global__ void k_wmma_gemmT(const float* __restrict__ A, const float* __restrict__ Bw,
                             const float* __restrict__ bias, float* __restrict__ C,
                             int M, int K, int act)
{
  constexpr bool NPAD = (NN % 16) != 0;
  const int lane = threadIdx.x & 31;
  const int half = lane >> 4;
  const int l16  = lane & 15;
  const int n0 = blockIdx.x * 64;
  const int m0 = blockIdx.y * 16;
  v8f acc[4] = {};
  const float2* Arow2 = (const float2*)(A + (size_t)(m0 + l16) * K);
  int nBt[4], nCt[4];
  const float* Bcol[4];
#pragma unroll
  for (int t = 0; t < 4; ++t) {
    int n = n0 + 16*t + l16;
    nBt[t] = n;
    nCt[t] = NPAD ? (n < NN ? n : NN - 1) : n;
    Bcol[t] = Bw + nCt[t];
  }
  const int Kmain = KPAD ? (K & ~31) : K;
  int kk = 0;
  for (; kk < Kmain; kk += 32) {
    v16h a;
#pragma unroll
    for (int j = 0; j < 8; ++j) {
      int ka = kk + koffA(j, half);
      float2 av = Arow2[ka >> 1];
      a[2*j]   = (_Float16)av.x;
      a[2*j+1] = (_Float16)av.y;
    }
#pragma unroll
    for (int t = 0; t < 4; ++t) {
      const float* Bb = Bcol[t] + (kk + 16*half) * NN;   // j-offsets are immediates
      v16h b;
#pragma unroll
      for (int j = 0; j < 8; ++j) {
        b[2*j]   = (_Float16)Bb[(2*j)     * NN];
        b[2*j+1] = (_Float16)Bb[(2*j + 1) * NN];
      }
      acc[t] = __builtin_amdgcn_wmma_f32_16x16x32_f16(false, a, false, b, (short)0, acc[t], false, false);
    }
  }
  if (KPAD) {                       // single tail iteration; K even -> pairs all-in or all-out
    v16h a;
#pragma unroll
    for (int j = 0; j < 8; ++j) {
      int ka = kk + koffA(j, half);
      bool in = ka < K;
      float2 av = Arow2[in ? (ka >> 1) : 0];
      a[2*j]   = (_Float16)(in ? av.x : 0.0f);
      a[2*j+1] = (_Float16)(in ? av.y : 0.0f);
    }
#pragma unroll
    for (int t = 0; t < 4; ++t) {
      v16h b;
#pragma unroll
      for (int j = 0; j < 8; ++j) {
        int kb = kk + koffB(j, half);
        int kb0 = kb < K ? kb : K - 1;     // B garbage multiplied by zeroed A
        int kb1 = kb + 1 < K ? kb + 1 : K - 1;
        b[2*j]   = (_Float16)Bcol[t][kb0 * NN];
        b[2*j+1] = (_Float16)Bcol[t][kb1 * NN];
      }
      acc[t] = __builtin_amdgcn_wmma_f32_16x16x32_f16(false, a, false, b, (short)0, acc[t], false, false);
    }
  }
#pragma unroll
  for (int t = 0; t < 4; ++t) {
    if (!NPAD || nBt[t] < NN) {
      float bv = bias ? bias[nCt[t]] : 0.0f;
#pragma unroll
      for (int r = 0; r < 8; ++r) {
        int m = m0 + r + 8*half;
        float v = acc[t][r] + bv;
        if (act == 1) v = v > 0.0f ? v : 0.0f;
        C[m * NN + nBt[t]] = v;
      }
    }
  }
}

// -------- fused FFN: Out = Hin + relu(Hin@W1 + b1)@W2 + b2 --------
// one block = 8 waves = one 16-row tile; hidden (16x512) staged in LDS as f16.
// DD (= dout) is compile-time: all W1/W2 fragment loads use immediate offsets.
// Stage 1: each wave owns 4 hidden tiles; A fragment reused across 4 WMMAs.
// REQUIRES M % 16 == 0.
template<int DD>
__global__ __launch_bounds__(256)
void k_ffn_fused(const float* __restrict__ Hin, const float* __restrict__ W1,
                 const float* __restrict__ b1, const float* __restrict__ W2,
                 const float* __restrict__ b2, float* __restrict__ Out, int M)
{
  constexpr bool DPAD = (DD % 32) != 0;
  __shared__ _Float16 hid[16 * FFNc];
  const int wave = threadIdx.x >> 5;
  const int lane = threadIdx.x & 31;
  const int half = lane >> 4;
  const int l16  = lane & 15;
  const int m0   = blockIdx.x * 16;
  const float2* Arow2 = (const float2*)(Hin + (size_t)(m0 + l16) * DD);

  // stage 1: hidden = relu(Hin_tile @ W1 + b1); K = DD, N = 512 (wave owns cols [wave*64, wave*64+64))
  {
    v8f acc[4] = {};
    const float* Bcol[4];
    int nBt[4];
#pragma unroll
    for (int t = 0; t < 4; ++t) {
      nBt[t] = wave * 64 + 16*t + l16;
      Bcol[t] = W1 + nBt[t];
    }
    constexpr int Kmain = DPAD ? (DD & ~31) : DD;
    int kk = 0;
    for (; kk < Kmain; kk += 32) {
      v16h a;
#pragma unroll
      for (int j = 0; j < 8; ++j) {
        int ka = kk + koffA(j, half);
        float2 av = Arow2[ka >> 1];
        a[2*j]   = (_Float16)av.x;
        a[2*j+1] = (_Float16)av.y;
      }
#pragma unroll
      for (int t = 0; t < 4; ++t) {
        const float* Bb = Bcol[t] + (kk + 16*half) * FFNc;
        v16h b;
#pragma unroll
        for (int j = 0; j < 8; ++j) {
          b[2*j]   = (_Float16)Bb[(2*j)     * FFNc];
          b[2*j+1] = (_Float16)Bb[(2*j + 1) * FFNc];
        }
        acc[t] = __builtin_amdgcn_wmma_f32_16x16x32_f16(false, a, false, b, (short)0, acc[t], false, false);
      }
    }
    if (DPAD) {
      v16h a;
#pragma unroll
      for (int j = 0; j < 8; ++j) {
        int ka = kk + koffA(j, half);
        bool in = ka < DD;
        float2 av = Arow2[in ? (ka >> 1) : 0];
        a[2*j]   = (_Float16)(in ? av.x : 0.0f);
        a[2*j+1] = (_Float16)(in ? av.y : 0.0f);
      }
#pragma unroll
      for (int t = 0; t < 4; ++t) {
        v16h b;
#pragma unroll
        for (int j = 0; j < 8; ++j) {
          int kb = kk + koffB(j, half);
          int kb0 = kb < DD ? kb : DD - 1;
          int kb1 = kb + 1 < DD ? kb + 1 : DD - 1;
          b[2*j]   = (_Float16)Bcol[t][kb0 * FFNc];
          b[2*j+1] = (_Float16)Bcol[t][kb1 * FFNc];
        }
        acc[t] = __builtin_amdgcn_wmma_f32_16x16x32_f16(false, a, false, b, (short)0, acc[t], false, false);
      }
    }
#pragma unroll
    for (int t = 0; t < 4; ++t) {
      float bv = b1[nBt[t]];
#pragma unroll
      for (int r = 0; r < 8; ++r) {
        float v = acc[t][r] + bv;
        hid[(r + 8*half) * FFNc + nBt[t]] = (_Float16)(v > 0.0f ? v : 0.0f);
      }
    }
  }
  __syncthreads();

  // stage 2: Out = Hin + hidden @ W2 + b2; K = 512 (aligned), N = DD
  constexpr int ntiles = (DD + 15) / 16;
  for (int nt = wave; nt < ntiles; nt += 8) {
    int nB = nt * 16 + l16;
    int nBc = DPAD ? (nB < DD ? nB : DD - 1) : nB;
    const float* Bcol = W2 + nBc;
    v8f acc = {};
    for (int kk = 0; kk < FFNc; kk += 32) {
      const float* Bb = Bcol + (kk + 16*half) * DD;
      v16h a, b;
#pragma unroll
      for (int j = 0; j < 8; ++j) {
        int ka = kk + koffA(j, half);
        h2v hv = *(const h2v*)&hid[l16 * FFNc + ka];
        a[2*j]   = hv[0];
        a[2*j+1] = hv[1];
        b[2*j]   = (_Float16)Bb[(2*j)     * DD];
        b[2*j+1] = (_Float16)Bb[(2*j + 1) * DD];
      }
      acc = __builtin_amdgcn_wmma_f32_16x16x32_f16(false, a, false, b, (short)0, acc, false, false);
    }
    float bv = b2[nBc];
#pragma unroll
    for (int r = 0; r < 8; ++r) {
      int m = m0 + r + 8*half;
      if (!DPAD || nB < DD)
        Out[m * DD + nB] = Hin[m * DD + nB] + acc[r] + bv;
    }
  }
}

// ---------------- elementwise / gather kernels (2-D grids, no int64 div) ------
__global__ void k_fill(float* p, long long n, float v) {
  long long i = (long long)blockIdx.x * blockDim.x + threadIdx.x;
  if (i < n) p[i] = v;
}

// grid = (rows, ceil(wd/256))
__global__ void k_gather_rows(const float* __restrict__ tab, const int* __restrict__ idx,
                              float* __restrict__ out, int wd) {
  int r = blockIdx.x;
  int c = blockIdx.y * blockDim.x + threadIdx.x;
  if (c >= wd) return;
  out[(size_t)r * wd + c] = tab[(size_t)idx[r] * wd + c];
}

__global__ void k_ngram(const float* __restrict__ embed, const int* __restrict__ sw,
                        float* __restrict__ out, int wd) {
  int s = blockIdx.x;
  int j = blockIdx.y * blockDim.x + threadIdx.x;
  if (j >= wd) return;
  float acc = 0.0f;
  for (int l = 0; l < Lc; ++l) acc += embed[(size_t)sw[s * Lc + l] * wd + j];
  out[(size_t)s * wd + j] = acc * (1.0f / (float)Lc);
}

__global__ void k_add_pos(const float* __restrict__ ng, const float* __restrict__ pos,
                          const int* __restrict__ sp, float* __restrict__ out, int wd) {
  int s = blockIdx.x;
  int j = blockIdx.y * blockDim.x + threadIdx.x;
  if (j >= wd) return;
  out[(size_t)s * wd + j] = ng[(size_t)s * wd + j] + pos[(size_t)sp[s] * wd + j];
}

__global__ void k_concat2(const float* __restrict__ a, const float* __restrict__ b,
                          float* __restrict__ out, int wa, int wb) {
  int r = blockIdx.x;
  int c = blockIdx.y * blockDim.x + threadIdx.x;
  int w = wa + wb;
  if (c >= w) return;
  out[(size_t)r * w + c] = (c < wa) ? a[(size_t)r * wa + c] : b[(size_t)r * wb + (c - wa)];
}

__global__ void k_copy_rows(const float* __restrict__ in, float* __restrict__ out,
                            int wd, int stride, int offset) {
  int r = blockIdx.x;
  int c = blockIdx.y * blockDim.x + threadIdx.x;
  if (c >= wd) return;
  out[(size_t)r * wd + c] = in[((size_t)r * stride + offset) * wd + c];
}

__global__ void k_add2(const float* __restrict__ a, const float* __restrict__ b,
                       float* __restrict__ out, long long n) {
  long long i = (long long)blockIdx.x * blockDim.x + threadIdx.x;
  if (i < n) out[i] = a[i] + b[i];
}

__global__ void k_elu(float* p, long long n) {
  long long i = (long long)blockIdx.x * blockDim.x + threadIdx.x;
  if (i >= n) return;
  float v = p[i];
  p[i] = v > 0.0f ? v : expm1f(v);
}

__global__ void k_iota_sp(int* src, int* dst, int n, int kshift) {
  int i = blockIdx.x * blockDim.x + threadIdx.x;
  if (i >= n) return;
  src[i] = i >> kshift; dst[i] = i;
}

// es[n,h] = dot(hs[n, h*dh : ], a_src[h, :]);  heads is a power of two
__global__ void k_es(const float* __restrict__ hs, const float* __restrict__ a_src,
                     float* __restrict__ es, int n, int hshift, int dh) {
  int i = blockIdx.x * blockDim.x + threadIdx.x;
  int heads = 1 << hshift;
  if (i >= (n << hshift)) return;
  int r = i >> hshift, h = i & (heads - 1);
  int dout = heads * dh;
  float acc = 0.0f;
  for (int k = 0; k < dh; ++k) acc += hs[(size_t)r * dout + h * dh + k] * a_src[h * dh + k];
  es[i] = acc;
}

// Wed[j,h] = sum_i Wdst[j, h*dh+i] * a_dst[h,i]
__global__ void k_wed(const float* __restrict__ Wdst, const float* __restrict__ a_dst,
                      float* __restrict__ wed, int ddst, int hshift, int dh) {
  int i = blockIdx.x * blockDim.x + threadIdx.x;
  int heads = 1 << hshift;
  if (i >= ddst * heads) return;
  int j = i >> hshift, h = i & (heads - 1);
  int dout = heads * dh;
  float acc = 0.0f;
  for (int k = 0; k < dh; ++k) acc += Wdst[(size_t)j * dout + h * dh + k] * a_dst[h * dh + k];
  wed[i] = acc;
}

// ebt[t,h] = sum_f tf_embed[t,f] * a_e[f,h]
__global__ void k_ebt(const float* __restrict__ tfe, const float* __restrict__ a_e,
                      float* __restrict__ ebt, int hshift) {
  int i = blockIdx.x * blockDim.x + threadIdx.x;
  int heads = 1 << hshift;
  if (i >= 10 * heads) return;
  int t = i >> hshift, h = i & (heads - 1);
  float acc = 0.0f;
  for (int f = 0; f < FEc; ++f) acc += tfe[t * FEc + f] * a_e[f * heads + h];
  ebt[i] = acc;
}

// one thread per edge; loops over heads (<= 8)
__global__ void k_edge_pre(const int* __restrict__ esrc, const int* __restrict__ edst,
                           const int* __restrict__ tf, const float* __restrict__ es,
                           const float* __restrict__ ed, const float* __restrict__ ebt,
                           float* __restrict__ e_buf, float* __restrict__ m,
                           int E, int heads) {
  int e = blockIdx.x * blockDim.x + threadIdx.x;
  if (e >= E) return;
  int s = esrc[e], d = edst[e];
  int t = tf ? tf[e] : 0;
  for (int h = 0; h < heads; ++h) {
    float v = es[s * heads + h] + ed[d * heads + h];
    if (ebt) v += ebt[t * heads + h];
    v = v > 0.0f ? v : 0.2f * v;               // leaky_relu(0.2)
    e_buf[e * heads + h] = v;
    atomicMaxF(&m[d * heads + h], v);
  }
}

__global__ void k_m_fix(float* m, long long n) {
  long long i = (long long)blockIdx.x * blockDim.x + threadIdx.x;
  if (i >= n) return;
  float v = m[i];
  if (!(v > -3.0e38f)) m[i] = 0.0f;            // isfinite(m) ? m : 0
}

__global__ void k_edge_exp(float* __restrict__ e_buf, const int* __restrict__ edst,
                           const float* __restrict__ m, float* __restrict__ s,
                           int E, int heads) {
  int e = blockIdx.x * blockDim.x + threadIdx.x;
  if (e >= E) return;
  int d = edst[e];
  for (int h = 0; h < heads; ++h) {
    float p = __expf(e_buf[e * heads + h] - m[d * heads + h]);
    e_buf[e * heads + h] = p;
    atomicAdd(&s[d * heads + h], p);
  }
}

// grid = (E, ceil(dout/256)); one thread per (edge, channel)
__global__ void k_edge_aggr(const float* __restrict__ e_buf, const float* __restrict__ s,
                            const float* __restrict__ hs, const int* __restrict__ esrc,
                            const int* __restrict__ edst, float* __restrict__ out,
                            int heads, int dh, int dout) {
  int e = blockIdx.x;
  int c = blockIdx.y * blockDim.x + threadIdx.x;
  if (c >= dout) return;
  int h = c / dh;
  int d = edst[e], sn = esrc[e];
  float alpha = e_buf[e * heads + h] / (s[d * heads + h] + 1e-9f);
  atomicAdd(&out[d * dout + c], hs[sn * dout + c] * alpha);
}

// ---------------- LSTM step (recurrent part) ----------------
__global__ __launch_bounds__(512)
void k_lstm_step(const float* __restrict__ xg, int xg_stride,
                 const float* __restrict__ Whh,
                 float* __restrict__ h, float* __restrict__ c,
                 float* __restrict__ hout, int hout_stride, int Hh)
{
  extern __shared__ float smem[];
  float* hs = smem;            // Hh
  float* gs = smem + Hh;       // 4*Hh
  int b = blockIdx.x, j = threadIdx.x;
  if (j < Hh) hs[j] = h[b * Hh + j];
  __syncthreads();
  float g = xg[b * xg_stride + j];
  for (int k = 0; k < Hh; ++k) g += hs[k] * Whh[k * 4 * Hh + j];
  gs[j] = g;
  __syncthreads();
  if (j < Hh) {
    float ig = gs[j], fg = gs[Hh + j], gg = gs[2 * Hh + j], og = gs[3 * Hh + j];
    float cn = dev_sigmoid(fg) * c[b * Hh + j] + dev_sigmoid(ig) * tanhf(gg);
    float hn = dev_sigmoid(og) * tanhf(cn);
    c[b * Hh + j] = cn;
    h[b * Hh + j] = hn;
    hout[b * hout_stride + j] = hn;
  }
}

// ---------------- doc attention & sd concat ----------------
__global__ void k_doc(const float* __restrict__ pz, const float* __restrict__ attl,
                      float* __restrict__ doc) {
  int b = blockIdx.x, j = threadIdx.x;   // blockDim = Hc
  float w[Pc]; float mx = -3.0e38f;
  for (int p = 0; p < Pc; ++p) {
    float v = attl[b * Pc + p];
    v = v > 0.0f ? v : 0.01f * v;        // leaky_relu(0.01)
    w[p] = v; mx = fmaxf(mx, v);
  }
  float sum = 0.0f;
  for (int p = 0; p < Pc; ++p) { w[p] = __expf(w[p] - mx); sum += w[p]; }
  float acc = 0.0f;
  for (int p = 0; p < Pc; ++p) acc += (w[p] / sum) * pz[(b * Pc + p) * Hc + j];
  doc[b * Hc + j] = acc;
}

// grid = (NS, 1); blockDim = 256 covers 2*Hc
__global__ void k_sd(const float* __restrict__ sent, const float* __restrict__ doc,
                     float* __restrict__ out) {
  int r = blockIdx.x;
  int c = threadIdx.x;                   // 0..255
  float v = (c < Hc) ? sent[r * Hc + c] : doc[(r >> 7) * Hc + (c - Hc)];
  out[r * 2 * Hc + c] = v;
}

// ---------------- host helpers ----------------
static inline int nb(long long n) { return (int)((n + 255) / 256); }
static inline dim3 g2(int rows, int wd) { return dim3((unsigned)rows, (unsigned)((wd + 255) / 256)); }
static inline int lg2(int x) { int s = 0; while ((1 << s) < x) ++s; return s; }

template<int NN>
static void gemm_n(hipStream_t st, const float* A, const float* W, const float* bias,
                   float* C, int M, int K, int act) {
  dim3 g((unsigned)((NN + 63) / 64), (unsigned)((M + 15) / 16));
  if ((K % 32) != 0) k_wmma_gemmT<NN, true ><<<g, 32, 0, st>>>(A, W, bias, C, M, K, act);
  else               k_wmma_gemmT<NN, false><<<g, 32, 0, st>>>(A, W, bias, C, M, K, act);
}

static void gemm(hipStream_t st, const float* A, const float* W, const float* bias,
                 float* C, int M, int K, int N, int act) {
  switch (N) {
    case 128: gemm_n<128>(st, A, W, bias, C, M, K, act); break;
    case 512: gemm_n<512>(st, A, W, bias, C, M, K, act); break;
    case 300: gemm_n<300>(st, A, W, bias, C, M, K, act); break;
    case 8:   gemm_n<8>  (st, A, W, bias, C, M, K, act); break;
    case 4:   gemm_n<4>  (st, A, W, bias, C, M, K, act); break;
    case 2:   gemm_n<2>  (st, A, W, bias, C, M, K, act); break;
    case 1:   gemm_n<1>  (st, A, W, bias, C, M, K, act); break;
    default:  break; // all call sites covered above
  }
}

static void ffn(hipStream_t st, const float* Hin, const float* W1, const float* b1,
                const float* W2, const float* b2, float* Out, int M, int Dd) {
  if (Dd == 300) k_ffn_fused<300><<<M / 16, 256, 0, st>>>(Hin, W1, b1, W2, b2, Out, M);
  else           k_ffn_fused<128><<<M / 16, 256, 0, st>>>(Hin, W1, b1, W2, b2, Out, M);
}

struct GatP {
  const float *Wsrc, *Wdst, *a_src, *a_dst, *W1, *b1, *W2, *b2, *a_e;
};

static void run_gat(hipStream_t st, const float* tf_embed,
                    const float* src, int n_src, int dsrc,
                    const float* dst, int n_dst, int ddst,
                    const GatP& p, int heads, int dout,
                    const int* esrc, const int* edst, const int* tf, int E,
                    float* hs, float* es, float* wed, float* ed, float* ebt,
                    float* m, float* s, float* out, float* res, float* e_buf)
{
  int dh = dout / heads;
  int hshift = lg2(heads);
  gemm(st, src, p.Wsrc, nullptr, hs, n_src, dsrc, dout, 0);
  k_es<<<nb((long long)n_src * heads), 256, 0, st>>>(hs, p.a_src, es, n_src, hshift, dh);
  k_wed<<<nb((long long)ddst * heads), 256, 0, st>>>(p.Wdst, p.a_dst, wed, ddst, hshift, dh);
  gemm(st, dst, wed, nullptr, ed, n_dst, ddst, heads, 0);
  if (p.a_e) k_ebt<<<nb(10 * heads), 256, 0, st>>>(tf_embed, p.a_e, ebt, hshift);
  long long nm = (long long)n_dst * heads;
  k_fill<<<nb(nm), 256, 0, st>>>(m, nm, -__builtin_inff());
  k_fill<<<nb(nm), 256, 0, st>>>(s, nm, 0.0f);
  k_fill<<<nb((long long)n_dst * dout), 256, 0, st>>>(out, (long long)n_dst * dout, 0.0f);
  k_edge_pre<<<nb(E), 256, 0, st>>>(esrc, edst, tf, es, ed, p.a_e ? ebt : nullptr, e_buf, m, E, heads);
  k_m_fix<<<nb(nm), 256, 0, st>>>(m, nm);
  k_edge_exp<<<nb(E), 256, 0, st>>>(e_buf, edst, m, s, E, heads);
  k_edge_aggr<<<g2(E, dout), 256, 0, st>>>(e_buf, s, hs, esrc, edst, out, heads, dh, dout);
  k_elu<<<nb((long long)n_dst * dout), 256, 0, st>>>(out, (long long)n_dst * dout);
  ffn(st, out, p.W1, p.b1, p.W2, p.b2, res, n_dst, dout);
}

extern "C" void kernel_launch(void* const* d_in, const int* in_sizes, int n_in,
                              void* d_out, int out_size, void* d_ws, size_t ws_size,
                              hipStream_t stream) {
  (void)in_sizes; (void)n_in; (void)out_size; (void)ws_size;
  auto F = [&](int i) { return (const float*)d_in[i]; };
  auto I = [&](int i) { return (const int*)d_in[i]; };

  const float* embed = F(0);
  const float* tf_embed = F(1);
  const float* pos = F(2);
  const float* cnnW = F(3); const float* cnnB = F(4);
  const float* ls_Wih_f = F(5); const float* ls_Whh_f = F(6); const float* ls_b_f = F(7);
  const float* ls_Wih_b = F(8); const float* ls_Whh_b = F(9); const float* ls_b_b = F(10);
  const float* lprojW = F(11); const float* lprojB = F(12);
  const float* nfeatW = F(13);
  const float* lp_Wih_f = F(14); const float* lp_Whh_f = F(15); const float* lp_b_f = F(16);
  const float* lp_Wih_b = F(17); const float* lp_Whh_b = F(18); const float* lp_b_b = F(19);
  const float* lppW = F(20); const float* lppB = F(21);
  GatP w2s { F(22), F(23), F(24), F(25), F(26), F(27), F(28), F(29), F(30) };
  GatP s2w { F(31), F(32), F(33), F(34), F(35), F(36), F(37), F(38), F(39) };
  GatP w2p { F(40), F(41), F(42), F(43), F(44), F(45), F(46), F(47), F(48) };
  GatP p2s { F(49), F(50), F(51), F(52), F(53), F(54), F(55), F(56), nullptr };
  const float* docW = F(57); const float* docB = F(58);
  const float* attW = F(59); const float* attB = F(60);
  const float* l1W = F(61);  const float* l1B = F(62);
  const float* whW = F(63);  const float* whB = F(64);
  const int* wid = I(65);
  const int* sent_words = I(66);
  const int* sent_pos = I(67);
  const int* ws_src = I(68); const int* ws_dst = I(69); const int* ws_tf = I(70);
  const int* wp_src = I(71); const int* wp_dst = I(72); const int* wp_tf = I(73);

  // ---- workspace bump allocation (floats) ----
  float* ws = (float*)d_ws;
  size_t off = 0;
  auto alloc = [&](size_t n) { float* p = ws + off; off += n; return p; };
  float* word_state = alloc((size_t)NWc * Dc);
  float* word_scr   = alloc((size_t)NWc * Dc);     // s2w aggr target / w2s,w2p hs
  float* e_buf      = alloc((size_t)E_WSc * 8);
  float* m_buf      = alloc((size_t)NWc * 4);
  float* s_buf      = alloc((size_t)NWc * 4);
  float* es_buf     = alloc((size_t)NWc * 8);
  float* ed_buf     = alloc((size_t)NWc * 4);
  float* wed_buf    = alloc(300 * 8);
  float* ebt_buf    = alloc(80);
  float* hs_sent    = alloc((size_t)NSc * Dc);
  float* hs_pass    = alloc((size_t)256 * Hc);
  float* sent_feat  = alloc((size_t)NSc * Hc);
  float* sent_state = alloc((size_t)NSc * Hc);
  float* s_w        = alloc((size_t)NSc * Hc);
  float* s_p        = alloc((size_t)NSc * Hc);
  float* sent_out   = alloc((size_t)NSc * Hc);
  float* pass_state = alloc((size_t)256 * Hc);
  float* pass_out   = alloc((size_t)256 * Hc);
  float* ngram      = alloc((size_t)NSc * Dc);
  float* cnn_in     = alloc((size_t)NSc * Dc);
  float* cnn_feat   = alloc((size_t)NSc * NFc);
  float* xg_f       = alloc((size_t)NSc * 4 * LHc);
  float* xg_b       = alloc((size_t)NSc * 4 * LHc);
  float* lstm_out   = alloc((size_t)NSc * 2 * LHc);
  float* lstm_feat  = alloc((size_t)NSc * NFc);
  float* cat_buf    = alloc((size_t)NSc * 2 * NFc);
  float* hcbuf      = alloc((size_t)4 * Bc * LHc);
  float* pxg_f      = alloc((size_t)NSc * 4 * Hc);
  float* pxg_b      = alloc((size_t)NSc * 4 * Hc);
  float* pseq       = alloc((size_t)NSc * 2 * Hc);
  float* phc        = alloc((size_t)2 * 256 * Hc);
  float* p_out      = alloc((size_t)256 * 2 * Hc);
  float* pz         = alloc((size_t)256 * Hc);
  float* attl       = alloc(256);
  float* doc        = alloc((size_t)Bc * Hc);
  float* l1_out     = alloc((size_t)NSc * Hc);
  int* sp_src       = (int*)alloc(NSc);
  int* sp_dst       = (int*)alloc(NSc);

  hipStream_t st = stream;

  // 1. word embeddings / n-grams / cnn features
  k_gather_rows<<<g2(NWc, Dc), 256, 0, st>>>(embed, wid, word_state, Dc);
  k_ngram<<<g2(NSc, Dc), 256, 0, st>>>(embed, sent_words, ngram, Dc);
  k_add_pos<<<g2(NSc, Dc), 256, 0, st>>>(ngram, pos, sent_pos, cnn_in, Dc);
  gemm(st, cnn_in, cnnW, cnnB, cnn_feat, NSc, Dc, NFc, 0);

  // 2. sentence BiLSTM (input projections hoisted into WMMA GEMMs)
  gemm(st, ngram, ls_Wih_f, ls_b_f, xg_f, NSc, Dc, 4 * LHc, 0);
  gemm(st, ngram, ls_Wih_b, ls_b_b, xg_b, NSc, Dc, 4 * LHc, 0);
  float* hS = hcbuf; float* cS = hcbuf + Bc * LHc;
  k_fill<<<nb(2 * Bc * LHc), 256, 0, st>>>(hS, 2 * Bc * LHc, 0.0f);
  for (int t = 0; t < S_DOCc; ++t)
    k_lstm_step<<<Bc, 4 * LHc, (LHc + 4 * LHc) * 4, st>>>(
        xg_f + (size_t)t * 4 * LHc, S_DOCc * 4 * LHc, ls_Whh_f, hS, cS,
        lstm_out + (size_t)t * 2 * LHc, S_DOCc * 2 * LHc, LHc);
  k_fill<<<nb(2 * Bc * LHc), 256, 0, st>>>(hS, 2 * Bc * LHc, 0.0f);
  for (int ss = 0; ss < S_DOCc; ++ss) {
    int t = S_DOCc - 1 - ss;
    k_lstm_step<<<Bc, 4 * LHc, (LHc + 4 * LHc) * 4, st>>>(
        xg_b + (size_t)t * 4 * LHc, S_DOCc * 4 * LHc, ls_Whh_b, hS, cS,
        lstm_out + (size_t)t * 2 * LHc + LHc, S_DOCc * 2 * LHc, LHc);
  }
  gemm(st, lstm_out, lprojW, lprojB, lstm_feat, NSc, 2 * LHc, NFc, 0);
  k_concat2<<<g2(NSc, 2 * NFc), 256, 0, st>>>(cnn_feat, lstm_feat, cat_buf, NFc, NFc);
  gemm(st, cat_buf, nfeatW, nullptr, sent_feat, NSc, 2 * NFc, Hc, 0);

  // 3. initial w2s GAT: words -> sentences
  run_gat(st, tf_embed, word_state, NWc, Dc, sent_feat, NSc, Hc, w2s, 8, Hc,
          ws_src, ws_dst, ws_tf, E_WSc, word_scr, es_buf, wed_buf, ed_buf, ebt_buf,
          m_buf, s_buf, sent_out, sent_state, e_buf);

  // 4. passage BiLSTM (batch = B*P = 256, seq = K = 16)
  gemm(st, sent_state, lp_Wih_f, lp_b_f, pxg_f, NSc, Hc, 4 * Hc, 0);
  gemm(st, sent_state, lp_Wih_b, lp_b_b, pxg_b, NSc, Hc, 4 * Hc, 0);
  float* hP = phc; float* cP = phc + 256 * Hc;
  k_fill<<<nb(2 * 256 * Hc), 256, 0, st>>>(hP, 2 * 256 * Hc, 0.0f);
  for (int t = 0; t < Kc; ++t)
    k_lstm_step<<<256, 4 * Hc, (Hc + 4 * Hc) * 4, st>>>(
        pxg_f + (size_t)t * 4 * Hc, Kc * 4 * Hc, lp_Whh_f, hP, cP,
        pseq + (size_t)t * 2 * Hc, Kc * 2 * Hc, Hc);
  k_fill<<<nb(2 * 256 * Hc), 256, 0, st>>>(hP, 2 * 256 * Hc, 0.0f);
  for (int ss = 0; ss < Kc; ++ss) {
    int t = Kc - 1 - ss;
    k_lstm_step<<<256, 4 * Hc, (Hc + 4 * Hc) * 4, st>>>(
        pxg_b + (size_t)t * 4 * Hc, Kc * 4 * Hc, lp_Whh_b, hP, cP,
        pseq + (size_t)t * 2 * Hc + Hc, Kc * 2 * Hc, Hc);
  }
  k_copy_rows<<<g2(256, 2 * Hc), 256, 0, st>>>(pseq, p_out, 2 * Hc, Kc, Kc - 1);
  gemm(st, p_out, lppW, lppB, pass_state, 256, 2 * Hc, Hc, 0);

  k_iota_sp<<<nb(NSc), 256, 0, st>>>(sp_src, sp_dst, NSc, 4 /* log2(Kc) */);

  // 5. message-passing iterations
  for (int it = 0; it < 2; ++it) {
    // s2w: sentences -> words (note reversed edge roles)
    run_gat(st, tf_embed, sent_state, NSc, Hc, word_state, NWc, Dc, s2w, 4, Dc,
            ws_dst, ws_src, ws_tf, E_WSc, hs_sent, es_buf, wed_buf, ed_buf, ebt_buf,
            m_buf, s_buf, word_scr, word_state, e_buf);
    // w2s: words -> sentences
    run_gat(st, tf_embed, word_state, NWc, Dc, sent_state, NSc, Hc, w2s, 8, Hc,
            ws_src, ws_dst, ws_tf, E_WSc, word_scr, es_buf, wed_buf, ed_buf, ebt_buf,
            m_buf, s_buf, sent_out, s_w, e_buf);
    // p2s: passages -> sentences
    run_gat(st, tf_embed, pass_state, 256, Hc, sent_state, NSc, Hc, p2s, 1, Hc,
            sp_src, sp_dst, nullptr, NSc, hs_pass, es_buf, wed_buf, ed_buf, ebt_buf,
            m_buf, s_buf, sent_out, s_p, e_buf);
    k_add2<<<nb((long long)NSc * Hc), 256, 0, st>>>(s_w, s_p, sent_state, (long long)NSc * Hc);
    // w2p: words -> passages
    run_gat(st, tf_embed, word_state, NWc, Dc, pass_state, 256, Hc, w2p, 1, Hc,
            wp_src, wp_dst, wp_tf, E_WPc, word_scr, es_buf, wed_buf, ed_buf, ebt_buf,
            m_buf, s_buf, pass_out, pass_state, e_buf);
  }

  // 6. doc attention + output head
  gemm(st, pass_state, docW, docB, pz, 256, Hc, Hc, 0);
  gemm(st, pz, attW, attB, attl, 256, Hc, 1, 0);
  k_doc<<<Bc, Hc, 0, st>>>(pz, attl, doc);
  k_sd<<<NSc, 256, 0, st>>>(sent_state, doc, cat_buf);
  gemm(st, cat_buf, l1W, l1B, l1_out, NSc, 2 * Hc, Hc, 1);
  gemm(st, l1_out, whW, whB, (float*)d_out, NSc, Hc, 2, 0);
}
